// BoundaryPredictor1_48189533061415
// MI455X (gfx1250) — matrix operand order repaired and checked by reading, not compiled, not run on hardware
//
#include <hip/hip_runtime.h>
#include <math.h>

#define B_ 8
#define L_ 4096
#define D_ 768
#define H_ 512
#define LDA 776      // padded LDS row stride (bf16 elems), multiple of 4
#define MT_ 4        // M-tiles (of 16 tokens) register-blocked per wave
#define TOK_WG (16 * MT_)   // 64 tokens per workgroup

typedef float  v8f   __attribute__((ext_vector_type(8)));
typedef float  v4f   __attribute__((ext_vector_type(4)));
typedef __bf16 v16bf __attribute__((ext_vector_type(16)));

union Frag {
  unsigned int u[8];
  uint4        q[2];
  v16bf        v;
};

__device__ __forceinline__ unsigned short f32_to_bf16(float f) {
  unsigned int x = __float_as_uint(f);
  x += 0x7FFFu + ((x >> 16) & 1u);   // round-to-nearest-even
  return (unsigned short)(x >> 16);
}

// ---------------------------------------------------------------------------
// Kernel 0: W1 [D][H] fp32 -> W1t [H][D] bf16 (N-major: B-frags load contiguously)
// ---------------------------------------------------------------------------
__global__ void __launch_bounds__(256) convert_w1_kernel(
    const float* __restrict__ W1, unsigned short* __restrict__ w1t) {
  int idx = blockIdx.x * 256 + threadIdx.x;          // over H_*D_ = 393216
  if (idx < H_ * D_) {
    int h = idx / D_;
    int d = idx - h * D_;
    w1t[idx] = f32_to_bf16(W1[d * H_ + h]);
  }
}

// ---------------------------------------------------------------------------
// Kernel Z: zero pooled output. Non-temporal stores: keep `hidden` resident
// in the 192MB L2 (it is re-read by pool_kernel) instead of caching this stream.
// ---------------------------------------------------------------------------
__global__ void __launch_bounds__(256) zero_pooled_kernel(float* __restrict__ out) {
  const size_t n4 = (size_t)B_ * L_ * D_ / 4;
  v4f* o = (v4f*)out;
  const v4f z = {0.f, 0.f, 0.f, 0.f};
  size_t stride = (size_t)gridDim.x * blockDim.x;
  for (size_t i = (size_t)blockIdx.x * blockDim.x + threadIdx.x; i < n4; i += stride)
    __builtin_nontemporal_store(z, &o[i]);
}

// ---------------------------------------------------------------------------
// Kernel 1: fused boundary MLP -> hard boundary bits
// grid = B*L/64 WGs, 256 threads (8 waves). WG owns 64 tokens (4 M-tiles held in
// registers); wave w owns N-tiles 4w..4w+3. Each B-fragment is reused by 4 WMMAs,
// cutting W1 L2 traffic 4x vs one-M-tile blocking.
// ---------------------------------------------------------------------------
__global__ void __launch_bounds__(256) logits_kernel(
    const float* __restrict__ hidden, const unsigned short* __restrict__ w1t,
    const float* __restrict__ b1, const float* __restrict__ W2,
    const float* __restrict__ b2, const float* __restrict__ noise_u,
    int* __restrict__ hard) {
  __shared__ unsigned short ldsA[TOK_WG * LDA];
  __shared__ float red[8][TOK_WG];

  const int tid = threadIdx.x;
  const long tokBase = (long)blockIdx.x * TOK_WG;   // flat over B*L (64 | 4096)

  // Stage 64xD fp32 tile -> bf16 LDS, vectorized (b128 loads, b64 LDS stores)
  {
    const float4* hsrc = (const float4*)(hidden + tokBase * D_);
    const int nvec = TOK_WG * D_ / 4;               // 12288
    for (int i = tid; i < nvec; i += 256) {
      int r  = i / (D_ / 4);
      int c4 = (i - r * (D_ / 4)) * 4;
      float4 v = hsrc[i];
      unsigned int lo = (unsigned int)f32_to_bf16(v.x) |
                        ((unsigned int)f32_to_bf16(v.y) << 16);
      unsigned int hi = (unsigned int)f32_to_bf16(v.z) |
                        ((unsigned int)f32_to_bf16(v.w) << 16);
      *(uint2*)&ldsA[r * LDA + c4] = make_uint2(lo, hi);
    }
  }
  __syncthreads();

  const int lane = tid & 31;
  const int wv   = tid >> 5;
  const int lh   = lane >> 4;    // lane half (0/1)
  const int ln   = lane & 15;    // A row (M) / B column (N) within tile

  int nIdx[4];
#pragma unroll
  for (int t = 0; t < 4; ++t) nIdx[t] = (wv * 4 + t) * 16 + ln;

  const v8f zero = {0.f, 0.f, 0.f, 0.f, 0.f, 0.f, 0.f, 0.f};
  v8f acc[MT_][4];
#pragma unroll
  for (int mt = 0; mt < MT_; ++mt)
#pragma unroll
    for (int t = 0; t < 4; ++t) acc[mt][t] = zero;

  for (int kb = 0; kb < D_ / 32; ++kb) {
    // A fragments (one per M-tile): lane row M=ln; elems 0..7 -> K=kb*32+lh*8+i,
    // elems 8..15 -> K=kb*32+16+lh*8+i  (ISA 16-bit A layout)
    Frag af[MT_];
#pragma unroll
    for (int mt = 0; mt < MT_; ++mt) {
      const uint4* a =
          (const uint4*)&ldsA[(mt * 16 + ln) * LDA + kb * 32 + lh * 8];
      af[mt].q[0] = a[0];
      af[mt].q[1] = *(const uint4*)&ldsA[(mt * 16 + ln) * LDA + kb * 32 + lh * 8 + 16];
    }
#pragma unroll
    for (int t = 0; t < 4; ++t) {
      // B fragment: lane column N=nIdx[t]; elems 0..7 -> K = kb*32 + lh*8 + i,
      // elems 8..15 -> K = kb*32 + 16 + lh*8 + i
      Frag bf;
      const uint4* bp =
          (const uint4*)&w1t[(size_t)nIdx[t] * D_ + kb * 32 + lh * 8];
      bf.q[0] = bp[0];
      bf.q[1] = bp[2];
#pragma unroll
      for (int mt = 0; mt < MT_; ++mt)
        acc[mt][t] = __builtin_amdgcn_wmma_f32_16x16x32_bf16(
            false, af[mt].v, false, bf.v, (short)0, acc[mt][t], false, false);
    }
  }

  // Epilogue: logits[m] = sum_n relu(acc + b1[n]) * W2[n]  (+ b2 later)
  float b1v[4], w2v[4];
#pragma unroll
  for (int t = 0; t < 4; ++t) {
    b1v[t] = b1[nIdx[t]];
    w2v[t] = W2[nIdx[t]];
  }
#pragma unroll
  for (int mt = 0; mt < MT_; ++mt) {
    float pm[8];
#pragma unroll
    for (int j = 0; j < 8; ++j) pm[j] = 0.f;
#pragma unroll
    for (int t = 0; t < 4; ++t) {
#pragma unroll
      for (int j = 0; j < 8; ++j) {
        float a = acc[mt][t][j] + b1v[t];   // C/D: VGPR j, lane -> M = j + 8*lh
        a = a > 0.f ? a : 0.f;
        pm[j] += a * w2v[t];
      }
    }
    // reduce across the 16 lanes (N within tile) of each half — deterministic
#pragma unroll
    for (int j = 0; j < 8; ++j) {
      for (int off = 1; off < 16; off <<= 1)
        pm[j] += __shfl_xor(pm[j], off, 32);
    }
    if (ln == 0) {
#pragma unroll
      for (int j = 0; j < 8; ++j) red[wv][mt * 16 + j + 8 * lh] = pm[j];
    }
  }
  __syncthreads();

  if (tid < TOK_WG) {
    float logit = b2[0];
#pragma unroll
    for (int w = 0; w < 8; ++w) logit += red[w][tid];
    float u = noise_u[tokBase + tid];
    float lgn = logf(u) - log1pf(-u);            // logistic noise
    // soft = sigmoid(logit + noise) > 0.5  <=>  logit + noise > 0
    hard[tokBase + tid] = (logit + lgn > 0.f) ? 1 : 0;
  }
}

// ---------------------------------------------------------------------------
// Kernel 2: per-batch exclusive scan of hard bits -> segment ids + batch counts
// ---------------------------------------------------------------------------
__global__ void __launch_bounds__(256) scan_kernel(
    const int* __restrict__ hard, int* __restrict__ seg, int* __restrict__ bcnt) {
  __shared__ int tot[256];
  const int b   = blockIdx.x;
  const int tid = threadIdx.x;
  const int base = b * L_ + tid * 16;

  int local[16];
  int s = 0;
#pragma unroll
  for (int i = 0; i < 16; ++i) {
    local[i] = s;
    s += hard[base + i];
  }
  tot[tid] = s;
  __syncthreads();
  if (tid == 0) {
    int acc = 0;
    for (int i = 0; i < 256; ++i) {
      int v = tot[i];
      tot[i] = acc;
      acc += v;
    }
    bcnt[b] = acc;
  }
  __syncthreads();
  int off = tot[tid];
#pragma unroll
  for (int i = 0; i < 16; ++i) seg[base + i] = off + local[i];
}

// ---------------------------------------------------------------------------
// Kernel 3: segment mean pooling. Segments are contiguous runs; one WG per token,
// only segment leaders do work (each hidden row read exactly once; hits L2 thanks
// to NT pooled stores). Deterministic, no atomics.
// ---------------------------------------------------------------------------
__global__ void __launch_bounds__(256) pool_kernel(
    const float* __restrict__ hidden, const int* __restrict__ hard,
    const int* __restrict__ seg, float* __restrict__ out) {
  const int g = blockIdx.x;
  const int b = g >> 12;          // / L_
  const int l = g & (L_ - 1);
  const int gidx = b * L_ + l;

  bool leader = (l == 0) || (hard[gidx - 1] == 1);
  if (!leader) return;            // uniform per block

  __shared__ int s_cnt;
  if (threadIdx.x == 0) {
    int j = l;
    while (j < L_ - 1 && hard[b * L_ + j] == 0) ++j;   // boundary or end of batch
    s_cnt = j - l + 1;
  }
  __syncthreads();
  const int cnt = s_cnt;
  const int sgid = seg[gidx];
  const float inv = 1.0f / (float)cnt;

  for (int d = threadIdx.x; d < D_; d += 256) {
    float sum = 0.f;
    for (int j = 0; j < cnt; ++j)
      sum += hidden[((long)gidx + j) * D_ + d];
    __builtin_nontemporal_store(sum * inv,
                                &out[((long)(b * L_ + sgid)) * D_ + d]);
  }
}

// ---------------------------------------------------------------------------
// Kernel 4: scalars (loss, num_boundaries, total_positions)
// ---------------------------------------------------------------------------
__global__ void finalize_kernel(const int* __restrict__ bcnt, float* __restrict__ outs) {
  int num = 0;
  for (int i = 0; i < B_; ++i) num += bcnt[i];
  const float total = (float)(B_ * L_);
  float ratio = (float)num / total;
  float loss  = fabsf(ratio - 0.25f) - 0.05f;   // PRIOR+0.05 = 0.25, margin 0.05
  loss = loss > 0.f ? loss : 0.f;
  outs[0] = loss;
  outs[1] = (float)num;
  outs[2] = total;
}

// ---------------------------------------------------------------------------
extern "C" void kernel_launch(void* const* d_in, const int* in_sizes, int n_in,
                              void* d_out, int out_size, void* d_ws, size_t ws_size,
                              hipStream_t stream) {
  const float* hidden = (const float*)d_in[0];   // [B,L,D]
  const float* W1     = (const float*)d_in[1];   // [D,H]
  const float* b1     = (const float*)d_in[2];   // [H]
  const float* W2     = (const float*)d_in[3];   // [H,1]
  const float* b2     = (const float*)d_in[4];   // [1]
  const float* noise  = (const float*)d_in[5];   // [B,L]
  float* out = (float*)d_out;

  char* ws = (char*)d_ws;
  unsigned short* w1t = (unsigned short*)ws;                       // 786432 B
  int* hard = (int*)(ws + 786432);                                 // 131072 B
  int* seg  = (int*)(ws + 786432 + 131072);                        // 131072 B
  int* bcnt = (int*)(ws + 786432 + 262144);                        // 32 B

  convert_w1_kernel<<<(H_ * D_ + 255) / 256, 256, 0, stream>>>(W1, w1t);
  zero_pooled_kernel<<<4096, 256, 0, stream>>>(out);
  logits_kernel<<<B_ * L_ / TOK_WG, 256, 0, stream>>>(hidden, w1t, b1, W2, b2,
                                                      noise, hard);
  scan_kernel<<<B_, 256, 0, stream>>>(hard, seg, bcnt);
  pool_kernel<<<B_ * L_, 256, 0, stream>>>(hidden, hard, seg, out);
  finalize_kernel<<<1, 1, 0, stream>>>(bcnt, out + (size_t)B_ * L_ * D_);
}
